// CrossAttention_seqz_76682346103304
// MI455X (gfx1250) — matrix operand order repaired and checked by reading, not compiled, not run on hardware
//
#include <hip/hip_runtime.h>
#include <hip/hip_bf16.h>

// Problem constants (match reference).
#define B_    4
#define LQ_   1024
#define LKV_  4096
#define C_    256
#define H_    8
#define Z_    1024
#define D_    32    // head dim = C/H
#define KVBLK_ 128  // kv rows staged per LDS buffer
#define NB_   (LKV_ / KVBLK_)

typedef __attribute__((ext_vector_type(16))) __bf16         v16bf;
typedef __attribute__((ext_vector_type(8)))  float          v8f;
typedef __attribute__((ext_vector_type(16))) unsigned short v16u;
typedef __attribute__((ext_vector_type(8)))  unsigned short v8u;
typedef __attribute__((ext_vector_type(4)))  unsigned int   v4u;
typedef __attribute__((ext_vector_type(8)))  int            v8i;
typedef __attribute__((ext_vector_type(4)))  int            v4i;

#if defined(__has_builtin)
#if __has_builtin(__builtin_amdgcn_tensor_load_to_lds)
#define HAVE_TDM 1
#endif
#endif
#ifndef HAVE_TDM
#define HAVE_TDM 0
#endif

static __device__ __forceinline__ unsigned short f2bf(float f) {
  unsigned u = __float_as_uint(f);
  u += 0x7FFFu + ((u >> 16) & 1u);       // round-to-nearest-even
  return (unsigned short)(u >> 16);
}

#if HAVE_TDM
// Issue one TDM 2D tile load (bf16 elements) global -> LDS.
// D# packing per CDNA5 ISA 8.3 (group0) / 8.4 (group1); groups 2/3 unused (2D).
static __device__ __forceinline__ void tdm_load_2d(
    const void* gptr, unsigned lds_addr,
    unsigned td0, unsigned td1,        // tensor dims (elements)
    unsigned tile0, unsigned tile1,    // tile dims   (elements)
    unsigned long long stride0)        // tensor dim0 stride (elements)
{
  unsigned long long ga = (unsigned long long)gptr;
  v4u g0;
  g0[0] = 1u;                                            // count=1, user desc
  g0[1] = lds_addr;                                      // lds_addr [63:32]
  g0[2] = (unsigned)(ga & 0xFFFFFFFFu);                  // global_addr lo
  g0[3] = (unsigned)((ga >> 32) & 0x01FFFFFFu)           // global_addr hi
        | (2u << 30);                                    // type = 2 ("image")
  v8i g1;
  g1[0] = (int)(1u << 16);                               // data_size=1 -> 2B
  g1[1] = (int)((td0 & 0xFFFFu) << 16);                  // tensor_dim0 lo16
  g1[2] = (int)(((td0 >> 16) & 0xFFFFu) | ((td1 & 0xFFFFu) << 16));
  g1[3] = (int)(((td1 >> 16) & 0xFFFFu) | ((tile0 & 0xFFFFu) << 16));
  g1[4] = (int)(tile1 & 0xFFFFu);                        // tile_dim2 = 0
  g1[5] = (int)(stride0 & 0xFFFFFFFFull);
  g1[6] = (int)((stride0 >> 32) & 0xFFFFull);            // dim1_stride = 0
  g1[7] = 0;
  v4i gz = {0, 0, 0, 0};
#if __clang_major__ >= 23
  v8i gz8 = {0, 0, 0, 0, 0, 0, 0, 0};
  __builtin_amdgcn_tensor_load_to_lds(g0, g1, gz, gz, gz8, 0);
#else
  __builtin_amdgcn_tensor_load_to_lds(g0, g1, gz, gz, 0);
#endif
}
#endif  // HAVE_TDM

// ---------------------------------------------------------------------------
// Prep kernels: fp32 -> bf16, APE folding.
// ---------------------------------------------------------------------------
__global__ void k_cvt_w(const float* __restrict__ w, unsigned short* __restrict__ o, int n) {
  int i = blockIdx.x * blockDim.x + threadIdx.x;
  if (i < n) o[i] = f2bf(w[i]);
}

__global__ void k_prep_q(const float* __restrict__ q, const float* __restrict__ q_ape,
                         unsigned short* __restrict__ o, int n) {
  int i = blockIdx.x * blockDim.x + threadIdx.x;
  if (i >= n) return;
  int lc = i % (LQ_ * C_);
  o[i] = f2bf(q[i] + q_ape[lc]);
}

__global__ void k_prep_kv(const float* __restrict__ kv, const float* __restrict__ k_ape,
                          unsigned short* __restrict__ o, int n) {
  int i = blockIdx.x * blockDim.x + threadIdx.x;
  if (i >= n) return;
  int lc = i % (LKV_ * C_);
  int l = lc / C_, c = lc % C_;
  float v = kv[i];
  if (l >= Z_) v += k_ape[(l - Z_) * C_ + c];   // aliases into both K and V paths
  o[i] = f2bf(v);
}

// ---------------------------------------------------------------------------
// Generic bf16 GEMM: Y = X @ W^T, one wave per 16x16 output tile.
//   MODE 0: bf16 head-major store   dst[((b*H+h)*L + lm)*D + dd]
//   MODE 1: bf16 head-transposed    dst[((b*H+h)*D + dd)*L + lm]   (for V)
//   MODE 2: f32 + bias              dst[m*C + n] = acc + bias[n]
// WMMA A/B fragments use contiguous k (16*half + e) in both operands; C/D layout per CDNA5 ISA 7.12.2.
// ---------------------------------------------------------------------------
template <int MODE>
__global__ void __launch_bounds__(32)
k_gemm(const unsigned short* __restrict__ X,
       const unsigned short* __restrict__ W,
       const float* __restrict__ bias,
       void* __restrict__ dst, int L) {
  const int tilesN = C_ / 16;
  int tile = blockIdx.x;
  int m0 = (tile / tilesN) * 16;
  int n0 = (tile % tilesN) * 16;
  int lane = threadIdx.x & 31;
  int r = lane & 15, half = lane >> 4;

  v8f acc = {};
#pragma unroll
  for (int k0 = 0; k0 < C_; k0 += 32) {
    // A fragment: row m0+r; element e -> K = 16*half + e
    const unsigned short* xr = X + (size_t)(m0 + r) * C_ + k0;
    v16u au = *(const v16u*)(xr + half * 16);
    // B fragment: B[k][n] = W[n][k]; lane -> N = n0+r, element e -> K = 16*half + e
    const unsigned short* wr = W + (size_t)(n0 + r) * C_ + k0 + half * 16;
    v16u bu = *(const v16u*)wr;
    acc = __builtin_amdgcn_wmma_f32_16x16x32_bf16(
        false, __builtin_bit_cast(v16bf, au),
        false, __builtin_bit_cast(v16bf, bu),
        (short)0, acc, false, false);
  }

  int n = n0 + r;
  int h = n / D_, dd = n % D_;
#pragma unroll
  for (int i = 0; i < 8; ++i) {        // C/D: VGPR i -> row m0 + i + 8*half
    int m = m0 + i + half * 8;
    int b = m / L, lm = m % L;
    float v = acc[i];
    if (MODE == 0)
      ((unsigned short*)dst)[(((size_t)b * H_ + h) * L + lm) * D_ + dd] = f2bf(v);
    else if (MODE == 1)
      ((unsigned short*)dst)[(((size_t)b * H_ + h) * D_ + dd) * L + lm] = f2bf(v);
    else
      ((float*)dst)[(size_t)m * C_ + n] = v + bias[n];
  }
}

// ---------------------------------------------------------------------------
// Flash attention, 4 waves / workgroup. Each workgroup owns (b, h, 64 q rows);
// wave w owns 16 of them. K/V kv-blocks of 128 are staged into double-buffered
// LDS by the Tensor Data Mover (wave 0 issues; s_wait_tensorcnt + barrier),
// software-pipelined two blocks deep. Online softmax in f32; P round-trips
// through a per-wave LDS tile (D-fragment -> A-fragment reshape).
// ---------------------------------------------------------------------------
__global__ void __launch_bounds__(128)
k_attn(const unsigned short* __restrict__ QH,   // (B,H,LQ,D)
       const unsigned short* __restrict__ KH,   // (B,H,LKV,D)
       const unsigned short* __restrict__ VT,   // (B,H,D,LKV)
       const float* __restrict__ attn_pos,      // (H,LQ,Z)
       unsigned short* __restrict__ XO) {       // (B,LQ,C) bf16
  __shared__ __align__(16) unsigned short Ks[2][KVBLK_ * D_];   // 128 x 32 row-major
  __shared__ __align__(16) unsigned short Vs[2][D_ * KVBLK_];   // 32 x 128 (dd-major)
  __shared__ __align__(32) unsigned short Pl[4][16 * 32];

  const int wave = threadIdx.x >> 5;
  const int lane = threadIdx.x & 31;
  const int r = lane & 15, half = lane >> 4;

  const int wg = blockIdx.x;                 // B*H*(LQ/64)
  const int qt = wg % (LQ_ / 64);
  const int bh = wg / (LQ_ / 64);
  const int h = bh % H_, b = bh / H_;
  const int q0 = qt * 64 + wave * 16;
  const float scale = 0.17677669529663687f;  // 1/sqrt(32)

  const unsigned short* Kbase = KH + (size_t)bh * LKV_ * D_;
  const unsigned short* Vbase = VT + (size_t)bh * D_ * LKV_;
  const float* Pbase = attn_pos + ((size_t)h * LQ_ + q0) * Z_;

  // Q A-fragment, loaded once: row q0+r, element e -> K = 16*half + e
  const unsigned short* qrow = QH + ((size_t)bh * LQ_ + q0 + r) * D_;
  v16u qu = *(const v16u*)(qrow + half * 16);
  const v16bf aQ = __builtin_bit_cast(v16bf, qu);

  v8f o0 = {}, o1 = {};
  float mrow[8], lrow[8];
#pragma unroll
  for (int i = 0; i < 8; ++i) { mrow[i] = -1e30f; lrow[i] = 0.f; }

#if HAVE_TDM
  // Prologue: stage blocks 0 and 1 (2 TDM ops per block, issued by wave 0).
  if (wave == 0) {
    tdm_load_2d(Kbase, (unsigned)(size_t)&Ks[0][0],
                KVBLK_ * D_, 1, KVBLK_ * D_, 1, KVBLK_ * D_);
    tdm_load_2d(Vbase, (unsigned)(size_t)&Vs[0][0],
                KVBLK_, D_, KVBLK_, D_, LKV_);
    tdm_load_2d(Kbase + (size_t)KVBLK_ * D_, (unsigned)(size_t)&Ks[1][0],
                KVBLK_ * D_, 1, KVBLK_ * D_, 1, KVBLK_ * D_);
    tdm_load_2d(Vbase + KVBLK_, (unsigned)(size_t)&Vs[1][0],
                KVBLK_, D_, KVBLK_, D_, LKV_);
  }
#endif

  for (int ib = 0; ib < NB_; ++ib) {
    const int buf = ib & 1;
    const int kblk = ib * KVBLK_;

#if HAVE_TDM
    if (wave == 0) {   // oldest outstanding block (= this buf) complete
      if (ib == NB_ - 1)
        __builtin_amdgcn_s_wait_tensorcnt(0);   // ImmArg: must be literal
      else
        __builtin_amdgcn_s_wait_tensorcnt(2);
    }
#else
    // Cooperative fallback copy: K contiguous, V row-by-row.
    for (int t = threadIdx.x; t < (KVBLK_ * D_) / 8; t += 128) {
      ((uint4*)&Ks[buf][0])[t] = ((const uint4*)(Kbase + (size_t)kblk * D_))[t];
      int dd = t >> 4, j = t & 15;
      ((uint4*)&Vs[buf][0])[t] =
          *(const uint4*)(Vbase + (size_t)dd * LKV_ + kblk + j * 8);
    }
#endif
    __syncthreads();

#pragma unroll
    for (int kk = 0; kk < KVBLK_ / 32; ++kk) {
      const int kg = kblk + kk * 32;

      // S = Q K^T from the LDS K tile (two 16x16 tiles, K-dim = d = 32)
      const unsigned short* Kt = &Ks[buf][(size_t)(kk * 32) * D_];
      v16u ku0 = *(const v16u*)(Kt + (size_t)r * D_ + half * 16);
      v16u ku1 = *(const v16u*)(Kt + (size_t)(16 + r) * D_ + half * 16);
      v8f zc = {};
      v8f s0 = __builtin_amdgcn_wmma_f32_16x16x32_bf16(
          false, aQ, false, __builtin_bit_cast(v16bf, ku0), (short)0, zc, false, false);
      v8f s1 = __builtin_amdgcn_wmma_f32_16x16x32_bf16(
          false, aQ, false, __builtin_bit_cast(v16bf, ku1), (short)0, zc, false, false);

#pragma unroll
      for (int i = 0; i < 8; ++i) { s0[i] *= scale; s1[i] *= scale; }
      if (kg < Z_) {   // positional bias on the template slice (uniform branch)
#pragma unroll
        for (int i = 0; i < 8; ++i) {
          int qr2 = i + half * 8;
          s0[i] += Pbase[(size_t)qr2 * Z_ + kg + r];
          s1[i] += Pbase[(size_t)qr2 * Z_ + kg + 16 + r];
        }
      }

      // Row max across the 16 lanes of this half (rows = i + 8*half).
      float tmax[8];
#pragma unroll
      for (int i = 0; i < 8; ++i) tmax[i] = fmaxf(s0[i], s1[i]);
#pragma unroll
      for (int off = 8; off >= 1; off >>= 1)
#pragma unroll
        for (int i = 0; i < 8; ++i) tmax[i] = fmaxf(tmax[i], __shfl_xor(tmax[i], off, 32));

      float alpha[8], tsum[8];
#pragma unroll
      for (int i = 0; i < 8; ++i) {
        float mnew = fmaxf(mrow[i], tmax[i]);
        alpha[i] = __expf(mrow[i] - mnew);
        mrow[i] = mnew;
        s0[i] = __expf(s0[i] - mnew);
        s1[i] = __expf(s1[i] - mnew);
        tsum[i] = s0[i] + s1[i];
      }
#pragma unroll
      for (int off = 8; off >= 1; off >>= 1)
#pragma unroll
        for (int i = 0; i < 8; ++i) tsum[i] += __shfl_xor(tsum[i], off, 32);
#pragma unroll
      for (int i = 0; i < 8; ++i) {
        lrow[i] = lrow[i] * alpha[i] + tsum[i];
        o0[i] *= alpha[i];
        o1[i] *= alpha[i];
      }

      // P: D-fragment -> per-wave LDS tile -> A-fragment (in-wave DS ordering).
#pragma unroll
      for (int i = 0; i < 8; ++i) {
        int row = i + half * 8;
        Pl[wave][row * 32 + r]      = f2bf(s0[i]);
        Pl[wave][row * 32 + 16 + r] = f2bf(s1[i]);
      }
      v16u pu = *(const v16u*)(&Pl[wave][r * 32] + half * 16);
      v16bf aP = __builtin_bit_cast(v16bf, pu);

      // O += P @ V from the LDS V tile (dd-major: contiguous B-fragments).
      const unsigned short* Vt = &Vs[buf][kk * 32];
      v16u vu0 = *(const v16u*)(Vt + (size_t)r * KVBLK_ + half * 16);
      v16u vu1 = *(const v16u*)(Vt + (size_t)(16 + r) * KVBLK_ + half * 16);
      o0 = __builtin_amdgcn_wmma_f32_16x16x32_bf16(
          false, aP, false, __builtin_bit_cast(v16bf, vu0), (short)0, o0, false, false);
      o1 = __builtin_amdgcn_wmma_f32_16x16x32_bf16(
          false, aP, false, __builtin_bit_cast(v16bf, vu1), (short)0, o1, false, false);
    }

    __syncthreads();   // everyone done reading buf before it is overwritten

#if HAVE_TDM
    if (wave == 0 && ib + 2 < NB_) {
      const int knext = kblk + 2 * KVBLK_;
      tdm_load_2d(Kbase + (size_t)knext * D_, (unsigned)(size_t)&Ks[buf][0],
                  KVBLK_ * D_, 1, KVBLK_ * D_, 1, KVBLK_ * D_);
      tdm_load_2d(Vbase + knext, (unsigned)(size_t)&Vs[buf][0],
                  KVBLK_, D_, KVBLK_, D_, LKV_);
    }
#endif
  }

  // Epilogue: x[b, q, h*32 + dd] = O / l  (bf16 for the final projection GEMM)
#pragma unroll
  for (int i = 0; i < 8; ++i) {
    int q = q0 + i + half * 8;
    float inv = 1.0f / lrow[i];
    size_t base = ((size_t)b * LQ_ + q) * C_ + h * D_;
    XO[base + r]      = f2bf(o0[i] * inv);
    XO[base + 16 + r] = f2bf(o1[i] * inv);
  }
}

// ---------------------------------------------------------------------------
extern "C" void kernel_launch(void* const* d_in, const int* in_sizes, int n_in,
                              void* d_out, int out_size, void* d_ws, size_t ws_size,
                              hipStream_t stream) {
  (void)in_sizes; (void)n_in; (void)out_size; (void)ws_size;
  const float* q        = (const float*)d_in[0];
  const float* kv       = (const float*)d_in[1];
  const float* q_ape    = (const float*)d_in[2];
  const float* k_ape    = (const float*)d_in[3];
  const float* attn_pos = (const float*)d_in[4];
  const float* Wq       = (const float*)d_in[5];
  const float* Wk       = (const float*)d_in[6];
  const float* Wv       = (const float*)d_in[7];
  const float* Wp       = (const float*)d_in[8];
  const float* bp       = (const float*)d_in[9];
  // d_in[10] = z_shape scalar; compile-time Z_ used.

  char* ws = (char*)d_ws;
  size_t off = 0;
  auto alloc = [&](size_t bytes) -> void* {
    void* p = ws + off;
    off += (bytes + 255) & ~(size_t)255;
    return p;
  };

  const size_t nW  = (size_t)C_ * C_;
  const size_t nQ  = (size_t)B_ * LQ_ * C_;
  const size_t nKV = (size_t)B_ * LKV_ * C_;

  unsigned short* WqB = (unsigned short*)alloc(nW * 2);
  unsigned short* WkB = (unsigned short*)alloc(nW * 2);
  unsigned short* WvB = (unsigned short*)alloc(nW * 2);
  unsigned short* WpB = (unsigned short*)alloc(nW * 2);
  unsigned short* QIN = (unsigned short*)alloc(nQ * 2);
  unsigned short* KVM = (unsigned short*)alloc(nKV * 2);
  unsigned short* QH  = (unsigned short*)alloc(nQ * 2);
  unsigned short* KH  = (unsigned short*)alloc(nKV * 2);
  unsigned short* VT  = (unsigned short*)alloc(nKV * 2);
  unsigned short* XO  = (unsigned short*)alloc(nQ * 2);

  // 1) precision prep
  k_cvt_w<<<(int)((nW + 255) / 256), 256, 0, stream>>>(Wq, WqB, (int)nW);
  k_cvt_w<<<(int)((nW + 255) / 256), 256, 0, stream>>>(Wk, WkB, (int)nW);
  k_cvt_w<<<(int)((nW + 255) / 256), 256, 0, stream>>>(Wv, WvB, (int)nW);
  k_cvt_w<<<(int)((nW + 255) / 256), 256, 0, stream>>>(Wp, WpB, (int)nW);
  k_prep_q<<<(int)((nQ + 255) / 256), 256, 0, stream>>>(q, q_ape, QIN, (int)nQ);
  k_prep_kv<<<(int)((nKV + 255) / 256), 256, 0, stream>>>(kv, k_ape, KVM, (int)nKV);

  // 2) projections (WMMA GEMMs)
  const int Mq = B_ * LQ_, Mkv = B_ * LKV_;
  k_gemm<0><<<(Mq / 16) * (C_ / 16), 32, 0, stream>>>(QIN, WqB, nullptr, QH, LQ_);
  k_gemm<0><<<(Mkv / 16) * (C_ / 16), 32, 0, stream>>>(KVM, WkB, nullptr, KH, LKV_);
  k_gemm<1><<<(Mkv / 16) * (C_ / 16), 32, 0, stream>>>(KVM, WvB, nullptr, VT, LKV_);

  // 3) flash attention (TDM-staged K/V, double-buffered)
  k_attn<<<B_ * H_ * (LQ_ / 64), 128, 0, stream>>>(QH, KH, VT, attn_pos, XO);

  // 4) output projection + bias (f32 out)
  k_gemm<2><<<(Mq / 16) * (C_ / 16), 32, 0, stream>>>(XO, WpB, bp, (float*)d_out, LQ_);
}
